// ContrastiveLoss_50079318671707
// MI455X (gfx1250) — matrix operand order repaired and checked by reading, not compiled, not run on hardware
//
#include <hip/hip_runtime.h>
#include <hip/hip_bf16.h>

// ---------------------------------------------------------------------------
// Supervised contrastive loss, fused for MI455X (gfx1250, wave32, WMMA).
//
// loss = (1/N) * sum_i [ n_i * log(sum_{j!=i} exp(sim_ij)) - sum_{j in pos_i} sim_ij ]
// where sim = normalize(E) @ normalize(E)^T / T, n_i = (#labels==label_i) - 1.
//
// GEMM part (8192x8192x128, 17.2 GFLOP) runs on v_wmma_f32_16x16x32_f16.
// Since B = E^T, the B tiles in WMMA layout are contiguous rows of the f16
// normalized matrix -> plain global_load_b128, no transpose ops needed.
// Dots are in [-1,1] so exp(sim) <= e^10: fp32-safe without max subtraction,
// enabling a single fused pass (never materializing the 268 MB sim matrix).
//
// Epilogue: branchless v_cndmask selects; exp is raw v_exp_f32 via
// __builtin_amdgcn_exp2f (arg bounded in [-14.5,14.5] -> no libm subnormal
// fixup needed); 1/T and log2(e) folded into one constant; positive-pair sum
// kept unscaled and multiplied by 1/T once at the end. The only data-dependent
// branch is the wave-uniform diagonal-tile test, forced scalar via
// readfirstlane (s_cmp/s_cbranch, no exec-mask traffic).
// ---------------------------------------------------------------------------

typedef __attribute__((ext_vector_type(16))) _Float16 v16h;
typedef __attribute__((ext_vector_type(8)))  _Float16 v8h;
typedef __attribute__((ext_vector_type(4)))  _Float16 v4h;
typedef __attribute__((ext_vector_type(8)))  float    v8f;

#define DIM      128
#define INV_T    10.0f
#define EXP_SCL  14.4269504088896340f   // (1/T) * log2(e): exp(c/T) = exp2(c*EXP_SCL)
#define JTILES   32                     // 32 j-tiles of 16 cols = 512-col strip/wave
#define JSTRIP   (JTILES * 16)

// ---- zero scratch accumulators (ws is poisoned 0xAA by harness) ------------
__global__ __launch_bounds__(256) void zero_ws_k(float* __restrict__ expsum,
                                                 float* __restrict__ possum,
                                                 int* __restrict__ hist, int n) {
    int i = blockIdx.x * blockDim.x + threadIdx.x;
    if (i < n) { expsum[i] = 0.0f; possum[i] = 0.0f; }
    if (i < 128) hist[i] = 0;
}

// ---- label histogram (n_i = hist[label_i]-1) -------------------------------
__global__ __launch_bounds__(256) void hist_k(const int* __restrict__ labels,
                                              int* __restrict__ hist, int n) {
    int i = blockIdx.x * blockDim.x + threadIdx.x;
    if (i < n) atomicAdd(&hist[labels[i]], 1);
}

// ---- L2-normalize rows, convert to f16 (one wave32 per 128-float row) ------
__global__ __launch_bounds__(256) void normalize_k(const float* __restrict__ emb,
                                                   _Float16* __restrict__ e16, int n) {
    int gid  = blockIdx.x * blockDim.x + threadIdx.x;
    int row  = gid >> 5;
    int lane = gid & 31;
    if (row >= n) return;
    float4 v = ((const float4*)(emb + (size_t)row * DIM))[lane];   // 32 lanes * 4 = 128
    float ss = fmaf(v.x, v.x, fmaf(v.y, v.y, fmaf(v.z, v.z, v.w * v.w)));
    ss += __shfl_xor(ss, 16); ss += __shfl_xor(ss, 8);
    ss += __shfl_xor(ss, 4);  ss += __shfl_xor(ss, 2); ss += __shfl_xor(ss, 1);
    float inv = 1.0f / fmaxf(sqrtf(ss), 1e-12f);
    v4h h;
    h[0] = (_Float16)(v.x * inv); h[1] = (_Float16)(v.y * inv);
    h[2] = (_Float16)(v.z * inv); h[3] = (_Float16)(v.w * inv);
    *(v4h*)(e16 + (size_t)row * DIM + lane * 4) = h;
}

// ---- fused sim-tile GEMM + row reductions ----------------------------------
// Grid: x = j-strips (N/512), y = i-blocks (N/128). Block = 8 waves; wave w
// owns rows [i0, i0+16) and sweeps 512 columns, one 16x16 WMMA tile at a time.
__global__ __launch_bounds__(256) void simloss_k(const _Float16* __restrict__ e16,
                                                 const int* __restrict__ labels,
                                                 float* __restrict__ expsum,
                                                 float* __restrict__ possum,
                                                 int n) {
    const int lane = threadIdx.x & 31;
    const int wave = threadIdx.x >> 5;
    const int half = lane >> 4;        // which 16-lane half
    const int nn   = lane & 15;        // tile column for C/B, tile row for A

    const int i0     = (blockIdx.y * 8 + wave) * 16;
    // i0 is wave-uniform by construction: pin it to an SGPR so the per-tile
    // diagonal test below compiles to s_cmp/s_cbranch (no exec-mask traffic).
    const int i0_s   = __builtin_amdgcn_readfirstlane(i0);
    const int j0base = blockIdx.x * JSTRIP;

    // A fragments for rows i0..i0+15, full K=128 (kept in regs for the strip).
    // A-layout 16x32 f16 (matching B): lane<16 holds K {k0..+15} of row nn;
    // lane>=16 holds K {k0+16..+31}.  -> two 16B contiguous loads.
    const _Float16* arow = e16 + (size_t)(i0 + nn) * DIM;
    v16h afrag[4];
#pragma unroll
    for (int kk = 0; kk < 4; ++kk) {
        const int k0 = kk * 32;
        v8h lo = *(const v8h*)(arow + k0 +     16 * half);
        v8h hi = *(const v8h*)(arow + k0 + 8 + 16 * half);
        afrag[kk] = __builtin_shufflevector(lo, hi, 0,1,2,3,4,5,6,7,
                                                    8,9,10,11,12,13,14,15);
    }

    // Each lane's C elements cover rows i0 + r + 8*half, r = 0..7.
    int labi[8];
#pragma unroll
    for (int r = 0; r < 8; ++r) labi[r] = labels[i0 + r + 8 * half];

    float expacc[8] = {0.f, 0.f, 0.f, 0.f, 0.f, 0.f, 0.f, 0.f};
    float posacc[8] = {0.f, 0.f, 0.f, 0.f, 0.f, 0.f, 0.f, 0.f};  // raw dot sums

#pragma unroll 2
    for (int t = 0; t < JTILES; ++t) {
        const int j0 = j0base + t * 16;
        // B-layout 32x16 f16: lane holds col nn, 16 contiguous K at 16*half.
        // B[k][n] = e16[j0+n][k]  -> contiguous 32B row chunk.
        const _Float16* brow = e16 + (size_t)(j0 + nn) * DIM + 16 * half;

        v8f c = {0.f, 0.f, 0.f, 0.f, 0.f, 0.f, 0.f, 0.f};
#pragma unroll
        for (int kk = 0; kk < 4; ++kk) {
            v16h b = *(const v16h*)(brow + kk * 32);
            c = __builtin_amdgcn_wmma_f32_16x16x32_f16(
                    false, afrag[kk], false, b, (short)0, c, false, false);
        }

        const int j    = j0 + nn;
        const int labj = labels[j];

        if (j0 != i0_s) {   // scalar branch: tile cannot touch the diagonal
#pragma unroll
            for (int r = 0; r < 8; ++r) {
                // raw v_exp_f32, no libm range fixup (arg in [-14.5, 14.5])
                expacc[r] += __builtin_amdgcn_exp2f(c[r] * EXP_SCL);
                posacc[r] += (labj == labi[r]) ? c[r] : 0.0f;   // v_cndmask
            }
        } else {
            // Diagonal tile (once per strip at most): also mask i == j.
#pragma unroll
            for (int r = 0; r < 8; ++r) {
                const int  i  = i0 + r + 8 * half;
                const bool nd = (i != j);
                float ev = __builtin_amdgcn_exp2f(c[r] * EXP_SCL);
                expacc[r] += nd ? ev : 0.0f;
                posacc[r] += (nd && labj == labi[r]) ? c[r] : 0.0f;
            }
        }
    }

    // Reduce across the 16 columns held by each lane-half (xor masks stay
    // inside the half), then lane nn==0 of each half commits 8 rows.
#pragma unroll
    for (int r = 0; r < 8; ++r) {
        float e = expacc[r], p = posacc[r];
        e += __shfl_xor(e, 1); e += __shfl_xor(e, 2);
        e += __shfl_xor(e, 4); e += __shfl_xor(e, 8);
        p += __shfl_xor(p, 1); p += __shfl_xor(p, 2);
        p += __shfl_xor(p, 4); p += __shfl_xor(p, 8);
        expacc[r] = e; posacc[r] = p;
    }
    if (nn == 0) {
#pragma unroll
        for (int r = 0; r < 8; ++r) {
            atomicAdd(&expsum[i0 + r + 8 * half], expacc[r]);
            atomicAdd(&possum[i0 + r + 8 * half], posacc[r]);
        }
    }
}

// ---- final scalar reduction ------------------------------------------------
__global__ __launch_bounds__(256) void finalize_k(const int* __restrict__ labels,
                                                  const int* __restrict__ hist,
                                                  const float* __restrict__ expsum,
                                                  const float* __restrict__ possum,
                                                  float* __restrict__ out, int n) {
    __shared__ float red[256];
    float acc = 0.0f;
    for (int i = threadIdx.x; i < n; i += 256) {
        float ni = (float)(hist[labels[i]] - 1);      // rows w/o positives: ni=0
        acc += ni * __logf(expsum[i]) - INV_T * possum[i];
    }
    red[threadIdx.x] = acc;
    __syncthreads();
    for (int s = 128; s > 0; s >>= 1) {
        if (threadIdx.x < (unsigned)s) red[threadIdx.x] += red[threadIdx.x + s];
        __syncthreads();
    }
    if (threadIdx.x == 0) out[0] = red[0] / (float)n;
}

extern "C" void kernel_launch(void* const* d_in, const int* in_sizes, int n_in,
                              void* d_out, int out_size, void* d_ws, size_t ws_size,
                              hipStream_t stream) {
    const float* emb    = (const float*)d_in[0];
    const int*   labels = (const int*)d_in[1];
    const int n = in_sizes[1];                         // 8192

    // workspace: [ e16 f16 N*128 | expsum f32 N | possum f32 N | hist i32 128 ]
    _Float16* e16    = (_Float16*)d_ws;
    float*    expsum = (float*)((char*)d_ws + (size_t)n * DIM * sizeof(_Float16));
    float*    possum = expsum + n;
    int*      hist   = (int*)(possum + n);

    zero_ws_k   <<<(n + 255) / 256, 256, 0, stream>>>(expsum, possum, hist, n);
    hist_k      <<<(n + 255) / 256, 256, 0, stream>>>(labels, hist, n);
    normalize_k <<<(n * 32 + 255) / 256, 256, 0, stream>>>(emb, e16, n);

    dim3 grid(n / JSTRIP, n / 128);                    // 16 x 64 blocks
    simloss_k   <<<grid, 256, 0, stream>>>(e16, labels, expsum, possum, n);

    finalize_k  <<<1, 256, 0, stream>>>(labels, hist, expsum, possum,
                                        (float*)d_out, n);
}
